// LSTM_81776177316029
// MI455X (gfx1250) — matrix operand order repaired and checked by reading, not compiled, not run on hardware
//
#include <hip/hip_runtime.h>
#include <hip/hip_bf16.h>
#include <stdint.h>

// ---------------------------------------------------------------------------
// LSTM LM forward, persistent-kernel version (single launch for recurrence).
// B=128, S=512, D=512, H=1024, O=64, V=50000
//
// - 64 GEMM blocks + 1 FC block, one timestep per device-wide barrier phase.
//   Barrier: 1 spinner/block + per-thread single acquire load (no spin storm).
// - GEMM block b owns 64 gate columns; its [W_hh;W_ih] B panel (193 KB) is
//   staged into LDS once (CDNA5: 320 KB/WGP) -> all 512 steps hit ds_load_b128
//   (row stride 772 dwords == 4 mod 64 banks: conflict-free).
// - FC block stages fc_w (128 KB) in LDS the same way (stride 516 dwords).
// - gx fused into recurrence: per-step K = [h_prev(1024) ; emb[x[b,s]](512)].
// - All GEMMs: v_wmma_f32_16x16x32_bf16, f32 accumulate; i/f/g/o in 4
//   accumulators of one wave so the c/h update never leaves registers.
// - Next step's embedding rows prefetched (global_prefetch_b8).
// ---------------------------------------------------------------------------

typedef __attribute__((ext_vector_type(16))) __bf16 v16bf;
typedef __attribute__((ext_vector_type(8)))  float  v8f;

#define Bn 128
#define Sn 512
#define Dn 512
#define Hn 1024
#define On 64
#define Vn 50000

#define GEMM_BLKS 64
#define NBLKS     65
#define KTOT      1536          // Hn + Dn
#define LDS_STRIDE 1544         // KTOT+8 -> 772 dwords (== 4 mod 64 banks)
#define FC_STRIDE  1032         // Hn+8   -> 516 dwords (== 4 mod 64 banks)
#define LDS_BYTES  (64 * LDS_STRIDE * 2)   // 197632 B (FC block uses 132 KB of it)

__device__ __forceinline__ v8f wmma_bf16(v16bf a, v16bf b, v8f c) {
  return __builtin_amdgcn_wmma_f32_16x16x32_bf16(
      false, a, false, b, (short)0, c, false, false);
}

// 16 bf16 for one lane as two 16B chunks (global_load_b128 / ds_load_b128 x2)
__device__ __forceinline__ v16bf load16(const __bf16* p, int o0, int o1) {
  union { uint4 q[2]; v16bf v; } u;
  u.q[0] = *reinterpret_cast<const uint4*>(p + o0);
  u.q[1] = *reinterpret_cast<const uint4*>(p + o1);
  return u.v;
}

__device__ __forceinline__ float sigmoid_f(float x) {
  return 1.0f / (1.0f + __expf(-x));
}
__device__ __forceinline__ float tanh_f(float x) {
  float e = __expf(2.0f * fabsf(x));          // inf-safe
  float t = 1.0f - 2.0f / (e + 1.0f);
  return copysignf(t, x);
}

// Device-wide split barrier. Release on arrival; one thread spins (relaxed);
// then EVERY thread does a single acquire load of the counter, which
// synchronizes with the release-RMW chain and invalidates its caches before
// the next step reads h/c produced by other blocks.
__device__ __forceinline__ void grid_barrier(unsigned* ctr, unsigned target) {
  __syncthreads();
  if (threadIdx.x == 0) {
    __hip_atomic_fetch_add(ctr, 1u, __ATOMIC_RELEASE, __HIP_MEMORY_SCOPE_AGENT);
    while (__hip_atomic_load(ctr, __ATOMIC_RELAXED, __HIP_MEMORY_SCOPE_AGENT) < target)
      __builtin_amdgcn_s_sleep(2);
  }
  __syncthreads();
  (void)__hip_atomic_load(ctr, __ATOMIC_ACQUIRE, __HIP_MEMORY_SCOPE_AGENT);
}

// -------------------------------- prep kernels ------------------------------

__global__ void k_to_bf16(const float* __restrict__ in, __bf16* __restrict__ out, int n) {
  int i = blockIdx.x * blockDim.x + threadIdx.x;
  int stride = gridDim.x * blockDim.x;
  for (; i < n; i += stride) out[i] = (__bf16)in[i];
}

__global__ void k_bias_sum(const float* __restrict__ b_ih, const float* __restrict__ b_hh,
                           float* __restrict__ bias, int n) {
  int i = blockIdx.x * blockDim.x + threadIdx.x;
  if (i < n) bias[i] = b_ih[i] + b_hh[i];
}

__global__ void k_zero_u32(uint32_t* __restrict__ p, int n) {
  int i = blockIdx.x * blockDim.x + threadIdx.x;
  int stride = gridDim.x * blockDim.x;
  for (; i < n; i += stride) p[i] = 0u;
}

// -------------------------------- persistent kernel -------------------------

__global__ __launch_bounds__(256) void k_lstm_persist(
    const int*    __restrict__ x,     // [B,S]
    const __bf16* __restrict__ embb,  // [V,D]
    const __bf16* __restrict__ whh,   // [4H,H]
    const __bf16* __restrict__ wih,   // [4H,D]
    const float*  __restrict__ bias,  // [4H] = b_ih + b_hh
    const __bf16* __restrict__ fcw,   // [O,H]
    const float*  __restrict__ fcb,   // [O]
    __bf16*       __restrict__ h0,    // [B,H] ping
    __bf16*       __restrict__ h1,    // [B,H] pong
    float*        __restrict__ cst,   // [B,H]
    unsigned*     __restrict__ bar,   // grid barrier counter (zeroed per call)
    float*        __restrict__ out)   // [B,S,O]
{
  extern __shared__ __align__(16) char smem_raw[];
  __bf16* ldsB = (__bf16*)smem_raw;

  const int blk  = blockIdx.x;
  const int lane = threadIdx.x & 31;
  const int wave = threadIdx.x >> 5;
  const int l16  = lane & 15;
  const bool hi  = lane >= 16;
  const int selA = hi ? 8 : 0;     // unused: A is loaded with selB too (K=16..31)
  const int selB = hi ? 16 : 0;    // B: lanes16-31 hold K=16..31

  // ---- one-time LDS staging ------------------------------------------------
  if (blk < GEMM_BLKS) {
    // LDS row r = g*16+c holds W row (g*H + blk*16 + c): 1024 from W_hh then
    // 512 from W_ih (dword copies, padded stride 772 dwords).
    const uint32_t* whh32 = (const uint32_t*)whh;
    const uint32_t* wih32 = (const uint32_t*)wih;
    uint32_t* lds32 = (uint32_t*)smem_raw;
    for (int idx = threadIdx.x; idx < 64 * (KTOT / 2); idx += 256) {
      int r  = idx / (KTOT / 2);
      int kd = idx - r * (KTOT / 2);
      int g = r >> 4, c = r & 15;
      int wrow = g * Hn + blk * 16 + c;
      uint32_t v = (kd < Hn / 2) ? whh32[(size_t)wrow * (Hn / 2) + kd]
                                 : wih32[(size_t)wrow * (Dn / 2) + (kd - Hn / 2)];
      lds32[r * (LDS_STRIDE / 2) + kd] = v;
    }
  } else {
    // FC block: stage fc_w [64 x 1024] with stride 516 dwords.
    const uint32_t* fcw32 = (const uint32_t*)fcw;
    uint32_t* lds32 = (uint32_t*)smem_raw;
    for (int idx = threadIdx.x; idx < 64 * (Hn / 2); idx += 256) {
      int r  = idx >> 9;            // /512
      int kd = idx & 511;
      lds32[r * (FC_STRIDE / 2) + kd] = fcw32[(size_t)r * (Hn / 2) + kd];
    }
  }
  __syncthreads();

  if (blk < GEMM_BLKS) {
    // ---------------- GEMM block: tn = blk, tm = wave ----------------------
    const int n0    = blk * 16 + l16;           // column within H
    const int arow  = wave * 16 + l16;          // batch row this lane feeds A
    const int mbase = wave * 16 + (hi ? 8 : 0);

    float bv[4];
    int   rb[4];                                // LDS row base (elements)
#pragma unroll
    for (int g = 0; g < 4; ++g) {
      bv[g] = bias[g * Hn + n0];
      rb[g] = (g * 16 + l16) * LDS_STRIDE;
    }

    for (int p = 0; p < Sn; ++p) {
      const __bf16* hin  = (p & 1) ? h1 : h0;
      __bf16*       hout = (p & 1) ? h0 : h1;

      v8f acc[4];
#pragma unroll
      for (int g = 0; g < 4; ++g)
        acc[g] = v8f{bv[g], bv[g], bv[g], bv[g], bv[g], bv[g], bv[g], bv[g]};

      const int token = x[arow * Sn + p];
      const __bf16* aH = hin  + (size_t)arow * Hn;
      const __bf16* aE = embb + (size_t)token * Dn;

      // prefetch next step's embedding row into this WGP's caches
      if (p + 1 < Sn) {
        const int tnext = x[arow * Sn + p + 1];
        const char* pf = (const char*)(embb + (size_t)tnext * Dn);
        __builtin_prefetch(pf,       0, 1);
        __builtin_prefetch(pf + 256, 0, 1);
        __builtin_prefetch(pf + 512, 0, 1);
        __builtin_prefetch(pf + 768, 0, 1);
      }

      // K over previous hidden state (B from LDS)
      for (int k = 0; k < Hn; k += 32) {
        v16bf a = load16(aH, k + selB, k + selB + 8);
#pragma unroll
        for (int g = 0; g < 4; ++g) {
          v16bf b = load16(ldsB + rb[g], k + selB, k + selB + 8);
          acc[g] = wmma_bf16(a, b, acc[g]);
        }
      }
      // K over gathered embedding row (B from LDS, offset Hn)
      for (int k = 0; k < Dn; k += 32) {
        v16bf a = load16(aE, k + selB, k + selB + 8);
#pragma unroll
        for (int g = 0; g < 4; ++g) {
          v16bf b = load16(ldsB + rb[g] + Hn, k + selB, k + selB + 8);
          acc[g] = wmma_bf16(a, b, acc[g]);
        }
      }

      // gate nonlinearities + state update, all in registers
#pragma unroll
      for (int v = 0; v < 8; ++v) {
        const int m = mbase + v;
        float iv = sigmoid_f(acc[0][v]);
        float fv = sigmoid_f(acc[1][v]);
        float gv = tanh_f(acc[2][v]);
        float ov = sigmoid_f(acc[3][v]);
        float cold = cst[m * Hn + n0];
        float cn = fv * cold + iv * gv;
        cst[m * Hn + n0] = cn;
        hout[m * Hn + n0] = (__bf16)(ov * tanh_f(cn));
      }

      grid_barrier(bar, (unsigned)(p + 1) * NBLKS);
    }
  } else {
    // ---------------- FC block: pipelined one step behind ------------------
    const int mbase = wave * 16 + (hi ? 8 : 0);
    const int arow  = wave * 16 + l16;
    float fb[4];
    int   rbF[4];
#pragma unroll
    for (int g = 0; g < 4; ++g) {
      fb[g]  = fcb[g * 16 + l16];
      rbF[g] = (g * 16 + l16) * FC_STRIDE;
    }

    for (int p = 0; p <= Sn; ++p) {
      if (p >= 1) {
        const int s = p - 1;
        const __bf16* hbuf = (s & 1) ? h0 : h1;   // where step s wrote h
        const __bf16* ap = hbuf + (size_t)arow * Hn;

        v8f acc[4];
#pragma unroll
        for (int g = 0; g < 4; ++g)
          acc[g] = v8f{fb[g], fb[g], fb[g], fb[g], fb[g], fb[g], fb[g], fb[g]};

        for (int k = 0; k < Hn; k += 32) {
          v16bf a = load16(ap, k + selB, k + selB + 8);
#pragma unroll
          for (int g = 0; g < 4; ++g) {
            v16bf b = load16(ldsB + rbF[g], k + selB, k + selB + 8);
            acc[g] = wmma_bf16(a, b, acc[g]);
          }
        }

        // log_softmax over O=64; xor masks 1..8 stay within 16-lane halves
#pragma unroll
        for (int v = 0; v < 8; ++v) {
          float mx = fmaxf(fmaxf(acc[0][v], acc[1][v]), fmaxf(acc[2][v], acc[3][v]));
#pragma unroll
          for (int d = 1; d < 16; d <<= 1) mx = fmaxf(mx, __shfl_xor(mx, d, 32));
          float sum = 0.0f;
#pragma unroll
          for (int g = 0; g < 4; ++g) sum += __expf(acc[g][v] - mx);
#pragma unroll
          for (int d = 1; d < 16; d <<= 1) sum += __shfl_xor(sum, d, 32);
          const float lse = mx + __logf(sum);
          const int m = mbase + v;
#pragma unroll
          for (int g = 0; g < 4; ++g)
            out[((size_t)m * Sn + s) * On + g * 16 + l16] = acc[g][v] - lse;
        }
      }
      if (p < Sn) grid_barrier(bar, (unsigned)(p + 1) * NBLKS);
    }
  }
}

// -------------------------------- launcher ----------------------------------

extern "C" void kernel_launch(void* const* d_in, const int* in_sizes, int n_in,
                              void* d_out, int out_size, void* d_ws, size_t ws_size,
                              hipStream_t stream) {
  const int*   x    = (const int*)  d_in[0];
  const float* emb  = (const float*)d_in[1];
  const float* W_ih = (const float*)d_in[2];
  const float* W_hh = (const float*)d_in[3];
  const float* b_ih = (const float*)d_in[4];
  const float* b_hh = (const float*)d_in[5];
  const float* fc_w = (const float*)d_in[6];
  const float* fc_b = (const float*)d_in[7];
  float* out = (float*)d_out;

  char* w = (char*)d_ws;
  size_t off = 0;
  auto alloc = [&](size_t bytes) {
    void* p = w + off;
    off += (bytes + 255) & ~(size_t)255;
    return p;
  };
  __bf16*   embb = (__bf16*)  alloc((size_t)Vn * Dn * 2);   // 51.2 MB
  __bf16*   wih  = (__bf16*)  alloc((size_t)4 * Hn * Dn * 2);
  __bf16*   whh  = (__bf16*)  alloc((size_t)4 * Hn * Hn * 2);
  __bf16*   fcw  = (__bf16*)  alloc((size_t)On * Hn * 2);
  float*    bias = (float*)   alloc((size_t)4 * Hn * 4);
  __bf16*   h0   = (__bf16*)  alloc((size_t)Bn * Hn * 2);
  __bf16*   h1   = (__bf16*)  alloc((size_t)Bn * Hn * 2);
  float*    cst  = (float*)   alloc((size_t)Bn * Hn * 4);
  unsigned* bar  = (unsigned*)alloc(256);
  (void)ws_size; (void)in_sizes; (void)n_in; (void)out_size;

  // 1) bf16 conversions + bias pre-sum
  k_to_bf16<<<2048, 256, 0, stream>>>(emb,  embb, Vn * Dn);
  k_to_bf16<<<1024, 256, 0, stream>>>(W_ih, wih,  4 * Hn * Dn);
  k_to_bf16<<<1024, 256, 0, stream>>>(W_hh, whh,  4 * Hn * Hn);
  k_to_bf16<<<64,   256, 0, stream>>>(fc_w, fcw,  On * Hn);
  k_bias_sum<<<16, 256, 0, stream>>>(b_ih, b_hh, bias, 4 * Hn);

  // 2) zero initial h, c, and the grid-barrier counter (ws is poisoned)
  k_zero_u32<<<64, 256, 0, stream>>>((uint32_t*)h0,  Bn * Hn / 2);
  k_zero_u32<<<64, 256, 0, stream>>>((uint32_t*)cst, Bn * Hn);
  k_zero_u32<<<1,  64,  0, stream>>>((uint32_t*)bar, 1);

  // 3) single persistent launch: 64 GEMM blocks + 1 FC block, 193 KB LDS each
  k_lstm_persist<<<NBLKS, 256, LDS_BYTES, stream>>>(
      x, embb, whh, wih, bias, fcw, fc_b, h0, h1, cst, bar, out);
}
